// HymbaBlock_42554535969386
// MI455X (gfx1250) — matrix operand order repaired and checked by reading, not compiled, not run on hardware
//
#include <hip/hip_runtime.h>
#include <hip/hip_bf16.h>
#include <math.h>

// ---------------------------------------------------------------------------
// Hymba block for MI455X (gfx1250, wave32, WMMA).
// All heavy matmuls use v_wmma_f32_16x16x32_bf16 (f32 accumulate).
// ---------------------------------------------------------------------------

typedef __attribute__((ext_vector_type(16))) __bf16 bf16x16;
typedef __attribute__((ext_vector_type(8)))  float  f32x8;

union FragAB {
  bf16x16 v;
  uint4   q[2];   // 32 bytes = 8 VGPRs
};

static __device__ __forceinline__ f32x8 wmma_bf16(bf16x16 a, bf16x16 b, f32x8 c) {
  // (neg_a, A, neg_b, B, c_mod, C, reuse_a, reuse_b)
  return __builtin_amdgcn_wmma_f32_16x16x32_bf16(false, a, false, b, (short)0, c,
                                                 false, false);
}

// ---------------- dims ----------------
#define LQ   2048   // sequence length
#define DMODEL 1024
#define IDIM 2048
#define NSSM 16
#define TSR_ 64
#define HQ   16
#define HKV_ 4
#define HDIM 128
#define WIN_ 1024
#define META_ 128

// ============================================================================
// GEMM: C[M,N] = A[M,K] @ B[K,N], f32 in/out, bf16 WMMA core.
// Block tile 256x128, BK=32, 256 threads = 8 waves; wave w -> rows [32w,32w+32)
// (two 16-row strips sharing B fragments -> 16 WMMAs / 20 ds_load_b128).
// Double-buffered LDS: global loads of tile t+1 overlap WMMA of tile t.
// Optional epilogue: +bias[col], softplus (for dt_proj).
// ============================================================================
#define BM 256
#define BN 128
#define BK 32

__global__ __launch_bounds__(256)
void gemm_bf16w(const float* __restrict__ A, const float* __restrict__ B,
                float* __restrict__ C, int M, int N, int K,
                const float* __restrict__ bias, int do_softplus)
{
  __shared__ __align__(16) __bf16 As[2][BM * BK];   // [m][k]   16 KB each
  __shared__ __align__(16) __bf16 Bs[2][BN * BK];   // [n][k]    8 KB each

  const int tid  = threadIdx.x;
  const int lane = tid & 31;
  const int wave = tid >> 5;
  const int half = lane >> 4;
  const int l16  = lane & 15;
  const int rowBase = blockIdx.y * BM;
  const int colBase = blockIdx.x * BN;
  const bool nEdge  = (colBase + BN) > N;     // only the x_proj tail tile

  f32x8 acc[2][8];
#pragma unroll
  for (int st = 0; st < 2; st++)
#pragma unroll
    for (int nt = 0; nt < 8; nt++)
#pragma unroll
      for (int j = 0; j < 8; j++) acc[st][nt][j] = 0.f;

  float4 aR[8];   // staged A tile: 256x32 f32 = 2048 float4, 8 per thread
  float4 bR[4];   // staged B tile:  32x128 f32 = 1024 float4, 4 per thread

  auto loadA = [&](int kk) {
#pragma unroll
    for (int i = 0; i < 8; i++) {
      int lin = tid + i * 256;
      int r   = lin >> 3;                 // 8 float4 per 32-elem row
      int c4  = (lin & 7) * 4;
      aR[i] = *(const float4*)(A + (size_t)(rowBase + r) * K + kk + c4);
    }
  };
  auto loadB = [&](int kk) {
#pragma unroll
    for (int i = 0; i < 4; i++) {
      int lin = tid + i * 256;
      int kr  = lin >> 5;                 // 32 float4 per 128-col row
      int n4  = (lin & 31) * 4;
      if (!nEdge) {
        bR[i] = *(const float4*)(B + (size_t)(kk + kr) * N + colBase + n4);
      } else {
        float t[4];
#pragma unroll
        for (int e = 0; e < 4; e++) {
          int col = colBase + n4 + e;
          t[e] = (col < N) ? B[(size_t)(kk + kr) * N + col] : 0.f;
        }
        bR[i] = make_float4(t[0], t[1], t[2], t[3]);
      }
    }
  };
  auto storeA = [&](int buf) {
#pragma unroll
    for (int i = 0; i < 8; i++) {
      int lin = tid + i * 256;
      int r   = lin >> 3;
      int c4  = (lin & 7) * 4;
      __bf16* dst = &As[buf][r * BK + c4];
      dst[0] = (__bf16)aR[i].x; dst[1] = (__bf16)aR[i].y;
      dst[2] = (__bf16)aR[i].z; dst[3] = (__bf16)aR[i].w;
    }
  };
  auto storeB = [&](int buf) {
#pragma unroll
    for (int i = 0; i < 4; i++) {
      int lin = tid + i * 256;
      int kr  = lin >> 5;
      int n4  = (lin & 31) * 4;
      Bs[buf][(n4 + 0) * BK + kr] = (__bf16)bR[i].x;
      Bs[buf][(n4 + 1) * BK + kr] = (__bf16)bR[i].y;
      Bs[buf][(n4 + 2) * BK + kr] = (__bf16)bR[i].z;
      Bs[buf][(n4 + 3) * BK + kr] = (__bf16)bR[i].w;
    }
  };

  // prologue: stage tile 0
  loadA(0); loadB(0);
  storeA(0); storeB(0);
  __syncthreads();

  for (int kk = 0; kk < K; kk += BK) {
    const int  buf     = (kk >> 5) & 1;
    const bool hasNext = (kk + BK) < K;
    if (hasNext) { loadA(kk + BK); loadB(kk + BK); }          // in-flight loads
    if (kk + 2 * BK < K)                                      // L2 pre-warm (t+2)
      __builtin_prefetch(A + (size_t)(rowBase + (tid >> 1)) * K + kk + 2 * BK, 0, 0);

    // ---- A fragments (two 16-row strips), exact 16x32 bf16 WMMA layout:
    // lane<16 -> K{0..7,16..23}; lane>=16 -> K{8..15,24..31}
    FragAB af[2];
    {
      const uint4* ap0 = (const uint4*)&As[buf][(wave * 32 + l16) * BK];
      af[0].q[0] = ap0[half];
      af[0].q[1] = ap0[2 + half];
      const uint4* ap1 = (const uint4*)&As[buf][(wave * 32 + 16 + l16) * BK];
      af[1].q[0] = ap1[half];
      af[1].q[1] = ap1[2 + half];
    }
#pragma unroll
    for (int nt = 0; nt < 8; nt++) {
      // B fragment (32x16): lane n holds col n; lane<16 -> K{0..7,16..23}, lane>=16 -> K{8..15,24..31}
      FragAB bfb;
      const uint4* bp = (const uint4*)&Bs[buf][(nt * 16 + l16) * BK];
      bfb.q[0] = bp[half];
      bfb.q[1] = bp[2 + half];
      acc[0][nt] = wmma_bf16(af[0].v, bfb.v, acc[0][nt]);
      acc[1][nt] = wmma_bf16(af[1].v, bfb.v, acc[1][nt]);
    }

    if (hasNext) { storeA(buf ^ 1); storeB(buf ^ 1); }        // waits loadcnt here
    __syncthreads();
  }

  // ---- epilogue + store (C layout: vgpr r -> row r (lanes<16) / r+8 (lanes>=16))
#pragma unroll
  for (int st = 0; st < 2; st++) {
    const int rbase = rowBase + wave * 32 + st * 16;
#pragma unroll
    for (int nt = 0; nt < 8; nt++) {
      int col = colBase + nt * 16 + l16;
      if (col >= N) continue;
      float b = bias ? bias[col] : 0.f;
#pragma unroll
      for (int r = 0; r < 8; r++) {
        int row = rbase + r + half * 8;
        float v = acc[st][nt][r] + b;
        if (do_softplus) v = (v > 20.f) ? v : log1pf(__expf(v));
        C[(size_t)row * N + col] = v;
      }
    }
  }
}

// ============================================================================
// Causal depthwise conv (K=4) + SiLU over latent = proj[:, :IDIM]
// ============================================================================
__global__ __launch_bounds__(256)
void conv_silu(const float* __restrict__ proj, const float* __restrict__ conv_w,
               const float* __restrict__ conv_b, float* __restrict__ u_cl)
{
  int id = blockIdx.x * 256 + threadIdx.x;        // LQ*IDIM threads
  int l = id >> 11;
  int i = id & (IDIM - 1);
  float acc = conv_b[i];
#pragma unroll
  for (int t = 0; t < 4; t++) {
    int ls = l - 3 + t;
    if (ls >= 0) acc += proj[(size_t)ls * (2 * IDIM) + i] * conv_w[i * 4 + t];
  }
  u_cl[(size_t)l * IDIM + i] = acc / (1.f + __expf(-acc));
}

// ============================================================================
// Per-row rmsnorms of ssm = [dt(64) | B(16) | C(16)]
// ============================================================================
__global__ __launch_bounds__(256)
void rms_ssm(const float* __restrict__ ssm, const float* __restrict__ dtw,
             const float* __restrict__ bw, const float* __restrict__ cw,
             float* __restrict__ dtn, float* __restrict__ Bn, float* __restrict__ Cn)
{
  int row = blockIdx.x * 256 + threadIdx.x;       // LQ rows
  const float* s = ssm + (size_t)row * (TSR_ + 2 * NSSM);
  float ms = 0.f;
  for (int j = 0; j < TSR_; j++) ms += s[j] * s[j];
  float r = rsqrtf(ms / TSR_ + 1e-6f);
  for (int j = 0; j < TSR_; j++) dtn[(size_t)row * TSR_ + j] = dtw[j] * s[j] * r;

  ms = 0.f;
  for (int j = 0; j < NSSM; j++) { float v = s[TSR_ + j]; ms += v * v; }
  r = rsqrtf(ms / NSSM + 1e-6f);
  for (int j = 0; j < NSSM; j++) Bn[(size_t)row * NSSM + j] = bw[j] * s[TSR_ + j] * r;

  ms = 0.f;
  for (int j = 0; j < NSSM; j++) { float v = s[TSR_ + NSSM + j]; ms += v * v; }
  r = rsqrtf(ms / NSSM + 1e-6f);
  for (int j = 0; j < NSSM; j++) Cn[(size_t)row * NSSM + j] = cw[j] * s[TSR_ + NSSM + j] * r;
}

// ============================================================================
// Selective-scan: one i-channel per thread, h[16] in registers, B/C staged in LDS.
// Fuses D-skip and SiLU(gate) multiply.
// ============================================================================
__global__ __launch_bounds__(256)
void ssm_scan(const float* __restrict__ dt2, const float* __restrict__ u_cl,
              const float* __restrict__ proj, const float* __restrict__ Bn,
              const float* __restrict__ Cn, const float* __restrict__ A_log,
              const float* __restrict__ D_skip, float* __restrict__ mamba)
{
  __shared__ float Bs[64 * NSSM];
  __shared__ float Cs[64 * NSSM];
  const int tid = threadIdx.x;
  const int i = blockIdx.x * 256 + tid;

  float Ai[NSSM];
#pragma unroll
  for (int n = 0; n < NSSM; n++) Ai[n] = -__expf(A_log[(size_t)i * NSSM + n]);
  const float Di = D_skip[i];
  float h[NSSM];
#pragma unroll
  for (int n = 0; n < NSSM; n++) h[n] = 0.f;

  for (int l0 = 0; l0 < LQ; l0 += 64) {
    for (int t = tid; t < 64 * NSSM; t += 256) {
      Bs[t] = Bn[(size_t)l0 * NSSM + t];
      Cs[t] = Cn[(size_t)l0 * NSSM + t];
    }
    __syncthreads();
    for (int lr = 0; lr < 64; lr++) {
      int l = l0 + lr;
      float dt = dt2[(size_t)l * IDIM + i];
      float u  = u_cl[(size_t)l * IDIM + i];
      float g  = proj[(size_t)l * (2 * IDIM) + IDIM + i];
      float y = 0.f;
#pragma unroll
      for (int n = 0; n < NSSM; n++) {
        float dA = __expf(dt * Ai[n]);
        h[n] = dA * h[n] + (dt * Bs[lr * NSSM + n]) * u;
        y += h[n] * Cs[lr * NSSM + n];
      }
      float out = y + u * Di;
      float sg = g / (1.f + __expf(-g));
      mamba[(size_t)l * IDIM + i] = out * sg;
    }
    __syncthreads();
  }
}

// ============================================================================
// RoPE + bf16 pack: qr[h][l][d], kr[h][l][d], vT[h][d][l] (V pre-transposed so
// attention B-fragments are contiguous 32B loads).
// ============================================================================
__global__ __launch_bounds__(256)
void rope_pack(const float* __restrict__ proj, const float* __restrict__ kbuf,
               const float* __restrict__ vbuf, __bf16* __restrict__ qr,
               __bf16* __restrict__ kr, __bf16* __restrict__ vT)
{
  int id = blockIdx.x * 256 + threadIdx.x;   // LQ * 3072
  int l = id / 3072;
  int c = id % 3072;
  if (c < IDIM) {                   // Q path
    int h = c >> 7, d = c & 127, dd = d & 63;
    float x = proj[(size_t)l * (2 * IDIM) + c];
    float invf = powf(10000.f, -(float)(2 * dd) / 128.f);
    float ang = (float)l * invf;
    float cs = cosf(ang), sn = sinf(ang);
    float other = (d < 64) ? -proj[(size_t)l * (2 * IDIM) + h * 128 + d + 64]
                           :  proj[(size_t)l * (2 * IDIM) + h * 128 + d - 64];
    qr[((size_t)h * LQ + l) * HDIM + d] = (__bf16)(x * cs + other * sn);
  } else if (c < IDIM + 512) {      // K path (HKV*HD = 512)
    int cc = c - IDIM, h = cc >> 7, d = cc & 127, dd = d & 63;
    float x = kbuf[(size_t)l * 512 + cc];
    float invf = powf(10000.f, -(float)(2 * dd) / 128.f);
    float ang = (float)l * invf;
    float cs = cosf(ang), sn = sinf(ang);
    float other = (d < 64) ? -kbuf[(size_t)l * 512 + h * 128 + d + 64]
                           :  kbuf[(size_t)l * 512 + h * 128 + d - 64];
    kr[((size_t)h * LQ + l) * HDIM + d] = (__bf16)(x * cs + other * sn);
  } else {                          // V path, transposed
    int cc = c - IDIM - 512, h = cc >> 7, d = cc & 127;
    vT[((size_t)h * HDIM + d) * LQ + l] = (__bf16)vbuf[(size_t)l * 512 + cc];
  }
}

// ============================================================================
// WMMA flash attention. Grid (L/16, H), 1 wave per block.
// 16 queries per block, 32 keys per step; mask = causal & (window | meta).
// ============================================================================
__global__ __launch_bounds__(32)
void flash_attn(const __bf16* __restrict__ qr, const __bf16* __restrict__ kr,
                const __bf16* __restrict__ vT, float* __restrict__ attn_o)
{
  __shared__ __align__(16) float  Sbuf[16 * 32];
  __shared__ __align__(16) __bf16 Pbuf[16 * 32];
  __shared__ float Abuf[16];
  __shared__ float Lbuf[16];

  const int lane = threadIdx.x;
  const int half = lane >> 4;
  const int l16  = lane & 15;
  const int h    = blockIdx.y;
  const int q0   = blockIdx.x * 16;
  const int kvh  = h >> 2;                        // repeat factor H/HKV = 4
  const float scale = 0.08838834764831845f;       // 1/sqrt(128)

  // Q fragments (4 k-chunks of 32 dims), A-layout straight from global
  FragAB qf[4];
  {
    const __bf16* qrow = qr + ((size_t)h * LQ + q0 + l16) * HDIM;
#pragma unroll
    for (int kc = 0; kc < 4; kc++) {
      const uint4* qp = (const uint4*)(qrow + kc * 32);
      qf[kc].q[0] = qp[half];
      qf[kc].q[1] = qp[2 + half];
    }
  }

  f32x8 o[8];
#pragma unroll
  for (int nt = 0; nt < 8; nt++)
#pragma unroll
    for (int j = 0; j < 8; j++) o[nt][j] = 0.f;
  float m_i = -3.0e38f, l_i = 0.f;                // per-row stats, lanes 0..15

  for (int kb = 0; kb < q0 + 16; kb += 32) {
    // skip tiles strictly between meta prefix and sliding window
    if (kb >= META_ && kb + 32 <= q0 - WIN_) continue;

    // ---- S = Q K^T (two 16-col tiles), mask, scale, spill to LDS ----
    for (int ct = 0; ct < 2; ct++) {
      f32x8 s;
#pragma unroll
      for (int j = 0; j < 8; j++) s[j] = 0.f;
      int key  = kb + ct * 16 + l16;
      int keyc = key < LQ ? key : (LQ - 1);       // clamp OOB (masked anyway)
      const __bf16* krow = kr + ((size_t)kvh * LQ + keyc) * HDIM;
#pragma unroll
      for (int kc = 0; kc < 4; kc++) {
        FragAB bfb;
        const uint4* bp = (const uint4*)(krow + kc * 32);
        bfb.q[0] = bp[half];
        bfb.q[1] = bp[2 + half];
        s = wmma_bf16(qf[kc].v, bfb.v, s);
      }
#pragma unroll
      for (int r = 0; r < 8; r++) {
        int row = r + half * 8;
        int qi = q0 + row;
        int ki = kb + ct * 16 + l16;
        bool valid = (qi >= ki) && (((qi - ki) <= WIN_) || (ki < META_));
        Sbuf[row * 32 + ct * 16 + l16] = valid ? s[r] * scale : -3.0e38f;
      }
    }
    __syncthreads();

    // ---- online-softmax row pass (lane r owns row r) ----
    if (lane < 16) {
      int r = lane;
      float mx = -3.0e38f;
      for (int j = 0; j < 32; j++) mx = fmaxf(mx, Sbuf[r * 32 + j]);
      float m_new = fmaxf(m_i, mx);
      float alpha = __expf(m_i - m_new);
      float sum = 0.f;
      for (int j = 0; j < 32; j++) {
        float sv = Sbuf[r * 32 + j];
        float p = (sv > -1.0e37f) ? __expf(sv - m_new) : 0.f;
        sum += p;
        Pbuf[r * 32 + j] = (__bf16)p;
      }
      l_i = l_i * alpha + sum;
      m_i = m_new;
      Abuf[r] = alpha;
    }
    __syncthreads();

    // ---- rescale O, then O += P @ V ----
    float ar[8];
#pragma unroll
    for (int r = 0; r < 8; r++) ar[r] = Abuf[r + half * 8];
#pragma unroll
    for (int nt = 0; nt < 8; nt++)
#pragma unroll
      for (int r = 0; r < 8; r++) o[nt][r] *= ar[r];

    FragAB pf;                                    // P in A-fragment layout
    {
      const uint4* pp = (const uint4*)Pbuf;       // row stride = 4 uint4
      pf.q[0] = pp[l16 * 4 + half];
      pf.q[1] = pp[l16 * 4 + 2 + half];
    }
#pragma unroll
    for (int nt = 0; nt < 8; nt++) {
      FragAB vf;                                  // B-frag from transposed V
      const uint4* vp = (const uint4*)(vT + ((size_t)kvh * HDIM + nt * 16 + l16) * LQ
                                       + kb);
      vf.q[0] = vp[half];
      vf.q[1] = vp[2 + half];
      o[nt] = wmma_bf16(pf.v, vf.v, o[nt]);
    }
    __syncthreads();
  }

  // ---- finalize: O /= l, write [L, I] with head-major columns ----
  if (lane < 16) Lbuf[lane] = 1.0f / l_i;
  __syncthreads();
  float lr[8];
#pragma unroll
  for (int r = 0; r < 8; r++) lr[r] = Lbuf[r + half * 8];
#pragma unroll
  for (int nt = 0; nt < 8; nt++)
#pragma unroll
    for (int r = 0; r < 8; r++) {
      int row = r + half * 8;
      attn_o[(size_t)(q0 + row) * IDIM + h * HDIM + nt * 16 + l16] = o[nt][r] * lr[r];
    }
}

// ============================================================================
// fused = 0.5*(rmsnorm(attn)*aw + rmsnorm(mamba)*mw), one block per row
// ============================================================================
__global__ __launch_bounds__(256)
void fuse_rms(const float* __restrict__ attn_o, const float* __restrict__ mamba,
              const float* __restrict__ aw, const float* __restrict__ mw,
              float* __restrict__ fused)
{
  __shared__ float r1[256], r2[256];
  const int row = blockIdx.x;
  const int tid = threadIdx.x;
  float sa = 0.f, sm = 0.f;
  for (int c = tid; c < IDIM; c += 256) {
    float a = attn_o[(size_t)row * IDIM + c]; sa += a * a;
    float m = mamba[(size_t)row * IDIM + c];  sm += m * m;
  }
  r1[tid] = sa; r2[tid] = sm;
  __syncthreads();
  for (int s = 128; s > 0; s >>= 1) {
    if (tid < s) { r1[tid] += r1[tid + s]; r2[tid] += r2[tid + s]; }
    __syncthreads();
  }
  float ra = rsqrtf(r1[0] / (float)IDIM + 1e-6f);
  float rm = rsqrtf(r2[0] / (float)IDIM + 1e-6f);
  for (int c = tid; c < IDIM; c += 256) {
    fused[(size_t)row * IDIM + c] =
        0.5f * (aw[c] * attn_o[(size_t)row * IDIM + c] * ra +
                mw[c] * mamba[(size_t)row * IDIM + c] * rm);
  }
}

// ============================================================================
// host-side launcher
// ============================================================================
extern "C" void kernel_launch(void* const* d_in, const int* in_sizes, int n_in,
                              void* d_out, int out_size, void* d_ws, size_t ws_size,
                              hipStream_t stream)
{
  (void)in_sizes; (void)n_in; (void)out_size; (void)ws_size;
  const float* x          = (const float*)d_in[0];
  const float* in_proj_w  = (const float*)d_in[1];
  const float* k_proj_w   = (const float*)d_in[2];
  const float* v_proj_w   = (const float*)d_in[3];
  const float* conv_w     = (const float*)d_in[4];
  const float* conv_b     = (const float*)d_in[5];
  const float* x_proj_w   = (const float*)d_in[6];
  const float* dt_proj_w  = (const float*)d_in[7];
  const float* dt_proj_b  = (const float*)d_in[8];
  const float* A_log      = (const float*)d_in[9];
  const float* D_skip     = (const float*)d_in[10];
  const float* dt_ln_w    = (const float*)d_in[11];
  const float* B_ln_w     = (const float*)d_in[12];
  const float* C_ln_w     = (const float*)d_in[13];
  const float* attn_ln_w  = (const float*)d_in[14];
  const float* mamba_ln_w = (const float*)d_in[15];
  const float* out_proj_w = (const float*)d_in[16];
  float* out = (float*)d_out;

  // ---- workspace carve-up (256B aligned) ----
  char* ws = (char*)d_ws;
  size_t off = 0;
  auto alloc = [&](size_t bytes) -> char* {
    char* p = ws + off;
    off += (bytes + 255) & ~(size_t)255;
    return p;
  };
  float* proj  = (float*)alloc((size_t)LQ * 2 * IDIM * 4);   // [L, 4096]
  float* kbuf  = (float*)alloc((size_t)LQ * 512 * 4);
  float* vbuf  = (float*)alloc((size_t)LQ * 512 * 4);
  float* u_cl  = (float*)alloc((size_t)LQ * IDIM * 4);       // reused as attn_o
  float* ssm   = (float*)alloc((size_t)LQ * (TSR_ + 2 * NSSM) * 4);
  float* dtn   = (float*)alloc((size_t)LQ * TSR_ * 4);
  float* Bn    = (float*)alloc((size_t)LQ * NSSM * 4);
  float* Cn    = (float*)alloc((size_t)LQ * NSSM * 4);
  float* dt2   = (float*)alloc((size_t)LQ * IDIM * 4);       // reused as fused
  float* mamba = (float*)alloc((size_t)LQ * IDIM * 4);
  __bf16* qr   = (__bf16*)alloc((size_t)HQ * LQ * HDIM * 2);
  __bf16* kr   = (__bf16*)alloc((size_t)HKV_ * LQ * HDIM * 2 + 256);
  __bf16* vT   = (__bf16*)alloc((size_t)HKV_ * HDIM * LQ * 2 + 256);
  float* attn_o = u_cl;   // u_cl dead after ssm_scan
  float* fused  = dt2;    // dt2 dead after ssm_scan

  const dim3 blk(256);

  // 1-3: projections
  gemm_bf16w<<<dim3(32, 8), blk, 0, stream>>>(x, in_proj_w, proj, LQ, 2 * IDIM, DMODEL, nullptr, 0);
  gemm_bf16w<<<dim3(4, 8),  blk, 0, stream>>>(x, k_proj_w,  kbuf, LQ, 512, DMODEL, nullptr, 0);
  gemm_bf16w<<<dim3(4, 8),  blk, 0, stream>>>(x, v_proj_w,  vbuf, LQ, 512, DMODEL, nullptr, 0);
  // 4: causal depthwise conv + SiLU
  conv_silu<<<(LQ * IDIM) / 256, blk, 0, stream>>>(proj, conv_w, conv_b, u_cl);
  // 5: x_proj (N=96, guarded)
  gemm_bf16w<<<dim3(1, 8), blk, 0, stream>>>(u_cl, x_proj_w, ssm, LQ, TSR_ + 2 * NSSM, IDIM, nullptr, 0);
  // 6: rmsnorms of dt/B/C
  rms_ssm<<<LQ / 256, blk, 0, stream>>>(ssm, dt_ln_w, B_ln_w, C_ln_w, dtn, Bn, Cn);
  // 7: dt_proj with fused bias + softplus epilogue
  gemm_bf16w<<<dim3(16, 8), blk, 0, stream>>>(dtn, dt_proj_w, dt2, LQ, IDIM, TSR_, dt_proj_b, 1);
  // 8: selective scan + D-skip + SiLU(gate)
  ssm_scan<<<IDIM / 256, blk, 0, stream>>>(dt2, u_cl, proj, Bn, Cn, A_log, D_skip, mamba);
  // 9: RoPE + bf16 packing (V transposed)
  rope_pack<<<(LQ * 3072) / 256, blk, 0, stream>>>(proj, kbuf, vbuf, qr, kr, vT);
  // 10: WMMA flash attention
  flash_attn<<<dim3(LQ / 16, HQ), dim3(32), 0, stream>>>(qr, kr, vT, attn_o);
  // 11: dual rmsnorm + average
  fuse_rms<<<LQ, blk, 0, stream>>>(attn_o, mamba, attn_ln_w, mamba_ln_w, fused);
  // 12: out_proj -> d_out
  gemm_bf16w<<<dim3(8, 8), blk, 0, stream>>>(fused, out_proj_w, out, LQ, DMODEL, IDIM, nullptr, 0);
}
